// MeshConv_45475113730482
// MI455X (gfx1250) — matrix operand order repaired and checked by reading, not compiled, not run on hardware
//
#include <hip/hip_runtime.h>

// ---------------------------------------------------------------------------
// MeshConv as a bf16 WMMA GEMM:  out[256,E] = W[256,640] x G[640,E] + bias
// G built on the fly from gathered neighbor features (self, sum, |diff| taps).
// Target: gfx1250 (wave32, v_wmma_f32_16x16x32_bf16, f32 accumulate).
//  - Kernel 1 (pack_w): convert W f32 -> bf16 pre-swizzled into the WMMA
//    16-bit A-fragment layout in d_ws (327 KB, L2 resident).
//  - Kernel 2 (GEMM): NT=64 edges/WG, 4 N-tiles x 2 M-tiles per wave; A
//    fragments are single 32B coalesced loads, B fragments from 80KB LDS.
// ---------------------------------------------------------------------------

typedef __attribute__((ext_vector_type(16))) __bf16 v16bf;
typedef __attribute__((ext_vector_type(8)))  float  v8f;

#define B_      4
#define CIN     128
#define COUT    256
#define E_      30000
#define RDIM    (CIN * 5)     // 640 reduction dim (c*5 + tap)
#define NT      64            // edges per workgroup (4 N-tiles of 16)
#define NTILES  (NT / 16)     // 4
#define KSTEPS  (RDIM / 32)   // 20 k-steps of K=32
#define MTILES  (COUT / 16)   // 16
#define WFRAGS  (MTILES * KSTEPS * 32)          // 10240 fragments-lanes
#define WPK_BYTES ((size_t)WFRAGS * 32)         // 327680 B in d_ws

__device__ __forceinline__ v16bf cvt_a_frag(const float* __restrict__ row, int kb) {
  // 16-bit A layout: this lane holds K = kb..kb+7 and kb+16..kb+23
  // (kb has the lane<16 ? 0 : 8 offset folded in by the caller).
  const float4 p0 = *(const float4*)(row + kb);
  const float4 p1 = *(const float4*)(row + kb + 4);
  const float4 p2 = *(const float4*)(row + kb + 16);
  const float4 p3 = *(const float4*)(row + kb + 20);
  v16bf a;
  a[0]  = (__bf16)p0.x; a[1]  = (__bf16)p0.y; a[2]  = (__bf16)p0.z; a[3]  = (__bf16)p0.w;
  a[4]  = (__bf16)p1.x; a[5]  = (__bf16)p1.y; a[6]  = (__bf16)p1.z; a[7]  = (__bf16)p1.w;
  a[8]  = (__bf16)p2.x; a[9]  = (__bf16)p2.y; a[10] = (__bf16)p2.z; a[11] = (__bf16)p2.w;
  a[12] = (__bf16)p3.x; a[13] = (__bf16)p3.y; a[14] = (__bf16)p3.z; a[15] = (__bf16)p3.w;
  return a;
}

// ---- pre-pack W into bf16 A-fragment layout: wpk[(m*KSTEPS+t)*32+lane] ----
__global__ __launch_bounds__(256)
void pack_w(const float* __restrict__ w, v16bf* __restrict__ wpk) {
  const int tid = blockIdx.x * blockDim.x + threadIdx.x;   // one frag-lane each
  if (tid >= WFRAGS) return;
  const int m    = tid / (KSTEPS * 32);
  const int rem  = tid - m * (KSTEPS * 32);
  const int t    = rem >> 5;
  const int lane = rem & 31;
  const int row  = m * 16 + (lane & 15);
  const int kb   = t * 32 + ((lane < 16) ? 0 : 8);
  wpk[tid] = cvt_a_frag(w + (long)row * RDIM, kb);
}

// ---- main GEMM ------------------------------------------------------------
template <bool PREPACKED>
__global__ __launch_bounds__(256)
void meshconv_wmma_bf16(const float* __restrict__ x,      // [B, CIN, E]
                        const int*   __restrict__ ne,     // [B, E, 4]
                        const float* __restrict__ w,      // [COUT, RDIM] f32
                        const v16bf* __restrict__ wpk,    // packed bf16 W
                        const float* __restrict__ bias,   // [COUT]
                        float*       __restrict__ out)    // [B, COUT, E]
{
  // G tile in WMMA-B fragment layout: [ntile][kstep][lane][16 bf16] = 80 KB
  __shared__ __attribute__((aligned(64))) __bf16 g_lds[RDIM * NT];
  __shared__ int idx_lds[NT * 4];

  const int tilesPerBatch = (E_ + NT - 1) / NT;
  const int b     = blockIdx.x / tilesPerBatch;
  const int ebase = (blockIdx.x % tilesPerBatch) * NT;
  const int tid   = threadIdx.x;
  const int lane  = tid & 31;
  const int wave  = tid >> 5;

  // ---- stage neighbor edge ids for this 64-edge tile --------------------
  {
    const int n = tid >> 2, j = tid & 3;           // 256 threads = NT*4 slots
    const int e = ebase + n;
    idx_lds[tid] = (e < E_) ? ne[((long)b * E_ + e) * 4 + j] : 0;
  }
  __syncthreads();

  // ---- build bf16 G tile in LDS (B-fragment layout) ---------------------
  {
    const int n     = tid & (NT - 1);              // edge column 0..63
    const int e     = ebase + n;
    const bool ev   = (e < E_);
    const int i1 = idx_lds[n * 4 + 0], i2 = idx_lds[n * 4 + 1];
    const int i3 = idx_lds[n * 4 + 2], i4 = idx_lds[n * 4 + 3];
    const int ntile = n >> 4, ncol = n & 15;

    #pragma unroll 4
    for (int i = 0; i < 32; ++i) {
      const int c = (tid >> 6) + 4 * i;            // channel 0..127
      const float* xc = x + ((long)b * CIN + c) * E_;
      const float v0 = ev ? xc[e]  : 0.f;
      const float v1 = ev ? xc[i1] : 0.f;
      const float v2 = ev ? xc[i2] : 0.f;
      const float v3 = ev ? xc[i3] : 0.f;
      const float v4 = ev ? xc[i4] : 0.f;
      float g[5];
      g[0] = v0;
      g[1] = v1 + v3;
      g[2] = v2 + v4;
      g[3] = fabsf(v1 - v3);
      g[4] = fabsf(v2 - v4);
      const int kb = c * 5;
      #pragma unroll
      for (int tp = 0; tp < 5; ++tp) {
        const int K  = kb + tp;                    // reduction index
        const int ts = K >> 5, kk = K & 31;
        const int fl = ncol + (((kk >> 3) & 1) << 4);    // fragment lane
        g_lds[((ntile * KSTEPS + ts) * 32 + fl) * 16 + ((kk & 7) | ((kk >> 4) << 3))] = (__bf16)g[tp];
      }
    }
  }
  __syncthreads();

  // ---- WMMA GEMM: each wave owns M-tiles {wave, wave+8} x N-tiles 0..3 --
  const int m0 = wave, m1 = wave + 8;
  const int koff = (lane < 16) ? 0 : 8;            // 16-bit A layout K offset
  const float* w0 = w + (long)(m0 * 16 + (lane & 15)) * RDIM;
  const float* w1 = w + (long)(m1 * 16 + (lane & 15)) * RDIM;

  v8f acc[2][NTILES] = {};

  for (int t = 0; t < KSTEPS; ++t) {
    v16bf a[2];
    if constexpr (PREPACKED) {
      a[0] = wpk[(m0 * KSTEPS + t) * 32 + lane];   // one 32B coalesced load
      a[1] = wpk[(m1 * KSTEPS + t) * 32 + lane];
      if (t + 1 < KSTEPS) {                        // keep next W frags warm
        __builtin_prefetch(&wpk[(m0 * KSTEPS + t + 1) * 32 + lane]);
        __builtin_prefetch(&wpk[(m1 * KSTEPS + t + 1) * 32 + lane]);
      }
    } else {
      const int kb = t * 32 + koff;
      a[0] = cvt_a_frag(w0, kb);
      a[1] = cvt_a_frag(w1, kb);
      if (t + 1 < KSTEPS) {
        __builtin_prefetch(w0 + kb + 32);
        __builtin_prefetch(w1 + kb + 32);
      }
    }

    v16bf bf[NTILES];
    #pragma unroll
    for (int nt = 0; nt < NTILES; ++nt)
      bf[nt] = *(const v16bf*)&g_lds[((nt * KSTEPS + t) * 32 + lane) * 16];

    #pragma unroll
    for (int mi = 0; mi < 2; ++mi)
      #pragma unroll
      for (int nt = 0; nt < NTILES; ++nt)
        acc[mi][nt] = __builtin_amdgcn_wmma_f32_16x16x32_bf16(
            false, a[mi], false, bf[nt], (short)0, acc[mi][nt], false, false);
  }

  // ---- bias + store (C/D layout: VGPR r -> M = r + 8*(lane>=16)) --------
  const int ecol  = ebase + (lane & 15);
  const int rbase = (lane >> 4) * 8;

  #pragma unroll
  for (int r = 0; r < 8; ++r) {
    const int o0 = m0 * 16 + rbase + r;
    const int o1 = m1 * 16 + rbase + r;
    const float b0 = bias[o0];
    const float b1 = bias[o1];
    #pragma unroll
    for (int nt = 0; nt < NTILES; ++nt) {
      const int e = ecol + nt * 16;
      if (e < E_) {
        out[((long)b * COUT + o0) * E_ + e] = acc[0][nt][r] + b0;
        out[((long)b * COUT + o1) * E_ + e] = acc[1][nt][r] + b1;
      }
    }
  }
}

extern "C" void kernel_launch(void* const* d_in, const int* in_sizes, int n_in,
                              void* d_out, int out_size, void* d_ws, size_t ws_size,
                              hipStream_t stream) {
  (void)in_sizes; (void)n_in; (void)out_size;
  const float* x    = (const float*)d_in[0];
  const int*   ne   = (const int*)d_in[1];
  const float* w    = (const float*)d_in[2];
  const float* bias = (const float*)d_in[3];
  float*       out  = (float*)d_out;

  const int tilesPerBatch = (E_ + NT - 1) / NT;   // 469
  dim3 grid(tilesPerBatch * B_);                  // 1876 workgroups
  dim3 block(256);                                // 8 waves (wave32)

  if (ws_size >= WPK_BYTES && d_ws != nullptr) {
    v16bf* wpk = (v16bf*)d_ws;
    pack_w<<<dim3((WFRAGS + 255) / 256), block, 0, stream>>>(w, wpk);
    meshconv_wmma_bf16<true><<<grid, block, 0, stream>>>(x, ne, w, wpk, bias, out);
  } else {
    meshconv_wmma_bf16<false><<<grid, block, 0, stream>>>(x, ne, w, nullptr, bias, out);
  }
}
